// RelativeAttention_3624952398113
// MI455X (gfx1250) — matrix operand order repaired and checked by reading, not compiled, not run on hardware
//
#include <hip/hip_runtime.h>
#include <math.h>

// ---------- types & helpers ----------
typedef __attribute__((ext_vector_type(16))) __bf16 bf16x16;
typedef __attribute__((ext_vector_type(8)))  __bf16 bf16x8;
typedef __attribute__((ext_vector_type(8)))  float  f32x8;

#define CAT16(lo, hi) __builtin_shufflevector((lo), (hi), 0,1,2,3,4,5,6,7,8,9,10,11,12,13,14,15)

__device__ __forceinline__ __bf16 f2bf(float f) {
  unsigned u = __float_as_uint(f);
  u = (u + 0x7FFFu + ((u >> 16) & 1u)) >> 16;
  unsigned short s = (unsigned short)u;
  return __builtin_bit_cast(__bf16, s);
}
__device__ __forceinline__ float bf2f(__bf16 b) {
  unsigned short s = __builtin_bit_cast(unsigned short, b);
  return __uint_as_float(((unsigned)s) << 16);
}
__device__ __forceinline__ f32x8 wmma_bf16(bf16x16 a, bf16x16 b, f32x8 c) {
  // v_wmma_f32_16x16x32_bf16  (D = A*B + C, fp32 accumulate)
  return __builtin_amdgcn_wmma_f32_16x16x32_bf16(false, a, false, b, (short)0, c, false, false);
}

#define NN    2
#define LL    1024
#define DD    1024
#define NHEAD 16
#define HDIM  64
#define NEG_BIG (-3.0e38f)

// ---------- fp32 -> bf16 bulk convert ----------
__global__ __launch_bounds__(256) void f2bf_kernel(const float* __restrict__ src,
                                                   __bf16* __restrict__ dst, int n) {
  int idx = blockIdx.x * 256 + threadIdx.x;
  if (idx < n) dst[idx] = f2bf(src[idx]);
}

// ---------- build R (sin/cos positional table), padded to 2048 rows ----------
__global__ __launch_bounds__(256) void rmat_kernel(__bf16* __restrict__ Rb) {
  int idx = blockIdx.x * 256 + threadIdx.x;
  if (idx >= 2048 * DD) return;
  int row = idx >> 10, col = idx & 1023;
  float val = 0.0f;
  if (row < 2 * LL - 1) {
    float p = (float)(row - (LL - 1));
    // inv_freq = 10000^(-(2c)/d) = exp2(-(2c/d)*log2(10000))
    float f = exp2f(-(float)(col >> 1) * (2.0f / (float)DD) * 13.2877123795494f);
    float ang = p * f;
    val = (col & 1) ? cosf(ang) : sinf(ang);
  }
  Rb[idx] = f2bf(val);
}

// ---------- tiled bf16 WMMA GEMM:  C[M,N] = A[M,K] @ B,  BM=BN=128, BK=32 ----
// 256 threads = 8 waves; wave (wm 0..3, wn 0..1) computes a 32x64 macro-tile as
// 2x4 WMMA tiles: 8 v_wmma per K-step from 2 A-frags + 4 B-frags.
// TRANSB=0: B is (K,N) row-major.  TRANSB=1: B is (N,K) row-major (use B^T).
// EPI=0: store bf16.  EPI=1: store f32 with +bias[col] +resid[row,col].
template <bool TRANSB, int EPI>
__global__ __launch_bounds__(256) void gemm_kernel(const __bf16* __restrict__ A,
                                                   const __bf16* __restrict__ B,
                                                   void* __restrict__ Cout,
                                                   const float* __restrict__ bias,
                                                   const float* __restrict__ resid,
                                                   int M, int Nn, int K) {
  __shared__ __align__(16) __bf16 As[128][40];   // [m][k]  (BK=32, padded)
  __shared__ __align__(16) __bf16 Bs[32][136];   // [k][n]  (BN=128, padded)
  const int tid  = threadIdx.x;
  const int wave = tid >> 5, lane = tid & 31;
  const int half = lane >> 4, lm = lane & 15;
  const int mb = blockIdx.y * 128, nb = blockIdx.x * 128;
  const int wm = (wave & 3) * 32;        // 4 m-groups of 32 rows
  const int wn = (wave >> 2) * 64;       // 2 n-groups of 64 cols

  f32x8 acc[2][4];
  for (int mt = 0; mt < 2; ++mt)
    for (int nt = 0; nt < 4; ++nt)
      for (int r = 0; r < 8; ++r) acc[mt][nt][r] = 0.0f;

  for (int k0 = 0; k0 < K; k0 += 32) {
    __syncthreads();
    for (int idx = tid; idx < 128 * 32; idx += 256) {
      int r = idx >> 5, c = idx & 31;
      As[r][c] = A[(size_t)(mb + r) * K + k0 + c];
    }
    for (int idx = tid; idx < 32 * 128; idx += 256) {
      int r = idx >> 7, c = idx & 127;
      Bs[r][c] = TRANSB ? B[(size_t)(nb + c) * K + k0 + r]
                        : B[(size_t)(k0 + r) * Nn + nb + c];
    }
    __syncthreads();

    // prefetch next K-tile into cache while WMMAs run (global_prefetch_b8)
    if (k0 + 32 < K) {
      __builtin_prefetch(&A[(size_t)(mb + (tid >> 1)) * K + k0 + 32 + (tid & 1) * 16], 0, 1);
      __builtin_prefetch(TRANSB ? &B[(size_t)(nb + (tid >> 1)) * K + k0 + 32 + (tid & 1) * 16]
                                : &B[(size_t)(k0 + 32 + (tid >> 3)) * Nn + nb + (tid & 7) * 16], 0, 1);
    }

    // A fragments: lane row M = lm, K runs [8*half..+7] and [16+8*half..+7]
    bf16x16 afrag[2];
    for (int mt = 0; mt < 2; ++mt) {
      const bf16x8 alo = *(const bf16x8*)&As[wm + mt * 16 + lm][half * 8];
      const bf16x8 ahi = *(const bf16x8*)&As[wm + mt * 16 + lm][16 + half * 8];
      afrag[mt] = CAT16(alo, ahi);
    }
    // B fragments: lane column N = lm, K runs [8*half..+7] and [16+8*half..+7]
    for (int nt = 0; nt < 4; ++nt) {
      bf16x16 bfrag;
      for (int e = 0; e < 16; ++e)
        bfrag[e] = Bs[(e & 7) + 8 * half + 16 * (e >> 3)][wn + nt * 16 + lm];
      for (int mt = 0; mt < 2; ++mt)
        acc[mt][nt] = wmma_bf16(afrag[mt], bfrag, acc[mt][nt]);
    }
  }

  for (int mt = 0; mt < 2; ++mt)
    for (int nt = 0; nt < 4; ++nt)
      for (int r = 0; r < 8; ++r) {
        int row = mb + wm + mt * 16 + r + 8 * half;
        int col = nb + wn + nt * 16 + lm;
        float v = acc[mt][nt][r];
        if (EPI == 0) {
          ((__bf16*)Cout)[(size_t)row * Nn + col] = f2bf(v);
        } else {
          ((float*)Cout)[(size_t)row * Nn + col] =
              v + bias[col] + resid[(size_t)row * Nn + col];
        }
      }
}

// ---------- per-key bias  Ac[n,h,j] = cb_h . k[n,h,j,:] ----------
__global__ __launch_bounds__(256) void acvec_kernel(const __bf16* __restrict__ kb,
                                                    const float* __restrict__ cb,
                                                    float* __restrict__ Ac) {
  int idx = blockIdx.x * 256 + threadIdx.x;
  if (idx >= NN * NHEAD * LL) return;
  int j = idx & (LL - 1);
  int h = (idx >> 10) & (NHEAD - 1);
  int n = idx >> 14;
  float s = 0.0f;
  for (int t = 0; t < HDIM; ++t)
    s += bf2f(kb[(size_t)(n * LL + j) * DD + h * HDIM + t]) * cb[h * HDIM + t];
  Ac[idx] = s;
}

// ---------- positional bias  Dv[h,r] = pb_h . Qm[r, h,:] ----------
__global__ __launch_bounds__(256) void dvec_kernel(const __bf16* __restrict__ Qmb,
                                                   const float* __restrict__ pb,
                                                   float* __restrict__ Dv) {
  int idx = blockIdx.x * 256 + threadIdx.x;
  if (idx >= NHEAD * 2048) return;
  int h = idx >> 11, r = idx & 2047;
  float s = 0.0f;
  if (r < 2 * LL - 1)
    for (int t = 0; t < HDIM; ++t)
      s += bf2f(Qmb[(size_t)r * DD + h * HDIM + t]) * pb[h * HDIM + t];
  Dv[idx] = s;
}

// ---------- fused causal attention with relative-position terms ----------
// grid.x = N*h, grid.y = L/64.  128 threads = 4 waves; wave w owns query rows
// [ibase+16w, ibase+16w+15].  Key loop over 32-wide key tiles (uniform count).
__global__ __launch_bounds__(128) void attn_kernel(const __bf16* __restrict__ qb,
                                                   const __bf16* __restrict__ kb,
                                                   const __bf16* __restrict__ vb,
                                                   const __bf16* __restrict__ Qmb,
                                                   const float* __restrict__ Ac,
                                                   const float* __restrict__ Dv,
                                                   __bf16* __restrict__ Ob) {
  const int nh = blockIdx.x;
  const int n = nh / NHEAD, h = nh % NHEAD;
  const int ibase = blockIdx.y * 64;
  const int tid = threadIdx.x;
  const int wave = tid >> 5, lane = tid & 31;
  const int half = lane >> 4, lm = lane & 15;
  const int qrow0 = ibase + wave * 16;

  __shared__ __align__(16) __bf16 Qs[64][72];     // q tile   [row][dim]
  __shared__ __align__(16) __bf16 Kst[64][40];    // k tile transposed [dim][j]
  __shared__ __align__(16) __bf16 Vs[32][72];     // v tile   [j][dim]
  __shared__ __align__(16) __bf16 Qwt[64][104];   // Qm union window transposed [dim][r]
  __shared__ __align__(16) __bf16 Pb[4][16][40];  // per-wave P round-trip
  __shared__ __align__(16) float  Wb[4][16][48];  // per-wave rel-B window

  // ---- stage q tile once ----
  for (int idx = tid; idx < 64 * 64; idx += 128) {
    int r = idx >> 6, c = idx & 63;
    Qs[r][c] = qb[(size_t)(n * LL + ibase + r) * DD + h * HDIM + c];
  }
  __syncthreads();

  // q A-fragments for both 32-wide K chunks (reused by score & rel-B GEMMs)
  bf16x16 qa[2];
  for (int kk = 0; kk < 2; ++kk) {
    const bf16x8 lo = *(const bf16x8*)&Qs[wave * 16 + lm][kk * 32 + half * 8];
    const bf16x8 hi = *(const bf16x8*)&Qs[wave * 16 + lm][kk * 32 + 16 + half * 8];
    qa[kk] = CAT16(lo, hi);
  }

  f32x8 Oacc[4];
  for (int t = 0; t < 4; ++t)
    for (int r = 0; r < 8; ++r) Oacc[t][r] = 0.0f;
  float mrow[8], lrow[8];
  for (int r = 0; r < 8; ++r) { mrow[r] = NEG_BIG; lrow[r] = 0.0f; }

  const int ntiles = (ibase >> 5) + 2;  // uniform across block
  for (int jt = 0; jt < ntiles; ++jt) {
    const int jbase = jt * 32;
    const int rbu = 960 - ibase + jbase;  // union window start row in Qm

    __syncthreads();
    for (int idx = tid; idx < 32 * 64; idx += 128) {
      int j = idx >> 6, c = idx & 63;
      size_t g = (size_t)(n * LL + jbase + j) * DD + h * HDIM + c;
      Kst[c][j] = kb[g];
      Vs[j][c] = vb[g];
    }
    for (int idx = tid; idx < 96 * 64; idx += 128) {
      int r = idx / 64, c = idx & 63;
      Qwt[c][r] = Qmb[(size_t)(rbu + r) * DD + h * HDIM + c];
    }
    __syncthreads();

    // prefetch next key/value tiles (global_prefetch_b8)
    if (jt + 1 < ntiles) {
      size_t gnext = (size_t)(n * LL + jbase + 32 + (tid >> 2)) * DD + h * HDIM + (tid & 3) * 16;
      __builtin_prefetch(&kb[gnext], 0, 1);
      __builtin_prefetch(&vb[gnext], 0, 1);
    }

    // ---- S = q @ k^T  (16x32 per wave: two 16x16 WMMA column tiles) ----
    f32x8 S[2];
    for (int s = 0; s < 2; ++s) {
      f32x8 a;
      for (int r = 0; r < 8; ++r) a[r] = 0.0f;
      for (int kk = 0; kk < 2; ++kk) {
        bf16x16 bk;
        for (int e = 0; e < 16; ++e)
          bk[e] = Kst[kk * 32 + (e & 7) + 8 * half + 16 * (e >> 3)][s * 16 + lm];
        a = wmma_bf16(qa[kk], bk, a);
      }
      S[s] = a;
    }

    // ---- relative window W[ii,t'] = q_ii . Qm[rb_w + t'],  t' in [0,48) ----
    for (int t = 0; t < 3; ++t) {
      f32x8 a;
      for (int r = 0; r < 8; ++r) a[r] = 0.0f;
      const int rloc = 48 - 16 * wave + 16 * t;  // offset into union window
      for (int kk = 0; kk < 2; ++kk) {
        bf16x16 bq;
        for (int e = 0; e < 16; ++e)
          bq[e] = Qwt[kk * 32 + (e & 7) + 8 * half + 16 * (e >> 3)][rloc + lm];
        a = wmma_bf16(qa[kk], bq, a);
      }
      for (int r = 0; r < 8; ++r) Wb[wave][r + 8 * half][t * 16 + lm] = a[r];
    }

    // ---- gather shifted rel term + biases, apply causal mask ----
    for (int s = 0; s < 2; ++s) {
      const int jj = s * 16 + lm;
      const int jglob = jbase + jj;
      const float acj = (jglob < LL) ? Ac[(n * NHEAD + h) * LL + jglob] : 0.0f;
      for (int r = 0; r < 8; ++r) {
        const int ii = r + 8 * half;
        const int iglob = qrow0 + ii;
        float sval = NEG_BIG;
        if (jglob <= iglob) {
          const int tp = jj - ii + 15;                 // 0..46
          const int rg = (LL - 1) + jglob - iglob;     // 0..1023
          sval = S[s][r] + Wb[wave][ii][tp] + acj + Dv[h * 2048 + rg];
        }
        S[s][r] = sval;
      }
    }

    // ---- online softmax (row reductions across the 16 lanes of each row) ----
    for (int r = 0; r < 8; ++r) {
      float mx = fmaxf(S[0][r], S[1][r]);
      mx = fmaxf(mx, __shfl_xor(mx, 1));
      mx = fmaxf(mx, __shfl_xor(mx, 2));
      mx = fmaxf(mx, __shfl_xor(mx, 4));
      mx = fmaxf(mx, __shfl_xor(mx, 8));
      const float mnew = fmaxf(mrow[r], mx);
      const float scale = expf(mrow[r] - mnew);
      float rs = 0.0f;
      for (int s = 0; s < 2; ++s) {
        float p = expf(S[s][r] - mnew);
        S[s][r] = p;
        rs += p;
      }
      rs += __shfl_xor(rs, 1);
      rs += __shfl_xor(rs, 2);
      rs += __shfl_xor(rs, 4);
      rs += __shfl_xor(rs, 8);
      lrow[r] = lrow[r] * scale + rs;
      mrow[r] = mnew;
      for (int t = 0; t < 4; ++t) Oacc[t][r] *= scale;
    }

    // ---- P (C-layout) -> LDS -> A-layout fragment, then O += P @ V ----
    for (int s = 0; s < 2; ++s)
      for (int r = 0; r < 8; ++r)
        Pb[wave][r + 8 * half][s * 16 + lm] = f2bf(S[s][r]);
    const bf16x8 plo = *(const bf16x8*)&Pb[wave][lm][half * 8];
    const bf16x8 phi = *(const bf16x8*)&Pb[wave][lm][16 + half * 8];
    bf16x16 pfrag = CAT16(plo, phi);
    for (int t = 0; t < 4; ++t) {
      bf16x16 bv;
      for (int e = 0; e < 16; ++e)
        bv[e] = Vs[(e & 7) + 8 * half + 16 * (e >> 3)][t * 16 + lm];
      Oacc[t] = wmma_bf16(pfrag, bv, Oacc[t]);
    }
  }

  // ---- finalize: O /= l, store bf16 at (n, i, h*H + d') ----
  for (int t = 0; t < 4; ++t)
    for (int r = 0; r < 8; ++r) {
      const int ii = r + 8 * half;
      const float o = Oacc[t][r] / lrow[r];
      Ob[(size_t)(n * LL + qrow0 + ii) * DD + h * HDIM + t * 16 + lm] = f2bf(o);
    }
}

// ---------- in-place LayerNorm over rows of d_out ----------
__global__ __launch_bounds__(256) void ln_kernel(float* __restrict__ Z,
                                                 const float* __restrict__ g,
                                                 const float* __restrict__ b) {
  const int row = blockIdx.x, tid = threadIdx.x;
  float v[4], s = 0.0f, s2 = 0.0f;
  for (int i = 0; i < 4; ++i) {
    v[i] = Z[(size_t)row * DD + tid + i * 256];
    s += v[i];
    s2 += v[i] * v[i];
  }
  for (int m = 16; m >= 1; m >>= 1) { s += __shfl_xor(s, m); s2 += __shfl_xor(s2, m); }
  __shared__ float ss[8], ss2[8];
  if ((tid & 31) == 0) { ss[tid >> 5] = s; ss2[tid >> 5] = s2; }
  __syncthreads();
  s = ss[tid & 7];
  s2 = ss2[tid & 7];
  for (int m = 4; m >= 1; m >>= 1) { s += __shfl_xor(s, m); s2 += __shfl_xor(s2, m); }
  const float mu = s * (1.0f / DD);
  const float var = s2 * (1.0f / DD) - mu * mu;
  const float inv = rsqrtf(var + 1e-5f);
  for (int i = 0; i < 4; ++i) {
    const int col = tid + i * 256;
    Z[(size_t)row * DD + col] = (v[i] - mu) * inv * g[col] + b[col];
  }
}

// ---------- host launcher ----------
extern "C" void kernel_launch(void* const* d_in, const int* in_sizes, int n_in,
                              void* d_out, int out_size, void* d_ws, size_t ws_size,
                              hipStream_t stream) {
  (void)in_sizes; (void)n_in; (void)out_size; (void)ws_size;
  const float* X   = (const float*)d_in[0];
  const float* Y   = (const float*)d_in[1];
  const float* Wq  = (const float*)d_in[4];
  const float* We  = (const float*)d_in[5];
  const float* Wv  = (const float*)d_in[6];
  const float* Wr  = (const float*)d_in[7];
  const float* cb  = (const float*)d_in[8];
  const float* pb  = (const float*)d_in[9];
  const float* Wo  = (const float*)d_in[10];
  const float* Wob = (const float*)d_in[11];
  const float* lng = (const float*)d_in[12];
  const float* lnb = (const float*)d_in[13];
  float* out = (float*)d_out;

  char* ws = (char*)d_ws;
  size_t off = 0;
  auto alloc = [&](size_t bytes) -> void* {
    void* p = ws + off;
    off += (bytes + 255) & ~(size_t)255;
    return p;
  };
  const size_t NLd = (size_t)NN * LL * DD;      // 2M elems
  const size_t dd  = (size_t)DD * DD;           // 1M elems
  __bf16* Xb  = (__bf16*)alloc(NLd * 2);
  __bf16* Yb  = (__bf16*)alloc(NLd * 2);
  __bf16* Wqb = (__bf16*)alloc(dd * 2);
  __bf16* Web = (__bf16*)alloc(dd * 2);
  __bf16* Wvb = (__bf16*)alloc(dd * 2);
  __bf16* Wrb = (__bf16*)alloc(dd * 2);
  __bf16* Wob16 = (__bf16*)alloc(dd * 2);
  __bf16* Rb  = (__bf16*)alloc((size_t)2048 * DD * 2);
  __bf16* qb  = (__bf16*)alloc(NLd * 2);
  __bf16* kb  = (__bf16*)alloc(NLd * 2);
  __bf16* vb  = (__bf16*)alloc(NLd * 2);
  __bf16* Qmb = (__bf16*)alloc((size_t)2048 * DD * 2);
  __bf16* Ob  = (__bf16*)alloc(NLd * 2);
  float*  Ac  = (float*)alloc((size_t)NN * NHEAD * LL * 4);
  float*  Dv  = (float*)alloc((size_t)NHEAD * 2048 * 4);

  // 1) fp32 -> bf16 conversions
  f2bf_kernel<<<(int)((NLd + 255) / 256), 256, 0, stream>>>(X, Xb, (int)NLd);
  f2bf_kernel<<<(int)((NLd + 255) / 256), 256, 0, stream>>>(Y, Yb, (int)NLd);
  f2bf_kernel<<<(int)((dd + 255) / 256), 256, 0, stream>>>(Wq, Wqb, (int)dd);
  f2bf_kernel<<<(int)((dd + 255) / 256), 256, 0, stream>>>(We, Web, (int)dd);
  f2bf_kernel<<<(int)((dd + 255) / 256), 256, 0, stream>>>(Wv, Wvb, (int)dd);
  f2bf_kernel<<<(int)((dd + 255) / 256), 256, 0, stream>>>(Wr, Wrb, (int)dd);
  f2bf_kernel<<<(int)((dd + 255) / 256), 256, 0, stream>>>(Wo, Wob16, (int)dd);

  // 2) positional table R
  rmat_kernel<<<(2048 * DD) / 256, 256, 0, stream>>>(Rb);

  // 3) WMMA GEMMs: projections + positional projection (bf16 out), 128x128 blocks
  dim3 gg(DD / 128, (NN * LL) / 128);  // (8, 16)
  gemm_kernel<false, 0><<<gg, 256, 0, stream>>>(Xb, Wqb, qb, nullptr, nullptr, NN * LL, DD, DD);
  gemm_kernel<false, 0><<<gg, 256, 0, stream>>>(Yb, Web, kb, nullptr, nullptr, NN * LL, DD, DD);
  gemm_kernel<false, 0><<<gg, 256, 0, stream>>>(Yb, Wvb, vb, nullptr, nullptr, NN * LL, DD, DD);
  gemm_kernel<false, 0><<<gg, 256, 0, stream>>>(Rb, Wrb, Qmb, nullptr, nullptr, 2048, DD, DD);

  // 4) bias vectors
  acvec_kernel<<<(NN * NHEAD * LL) / 256, 256, 0, stream>>>(kb, cb, Ac);
  dvec_kernel<<<(NHEAD * 2048) / 256, 256, 0, stream>>>(Qmb, pb, Dv);

  // 5) fused causal attention with relative shifts
  attn_kernel<<<dim3(NN * NHEAD, LL / 64), 128, 0, stream>>>(qb, kb, vb, Qmb, Ac, Dv, Ob);

  // 6) output projection (B transposed) + bias + residual -> fp32 d_out
  gemm_kernel<true, 1><<<gg, 256, 0, stream>>>(Ob, Wob16, out, Wob, X, NN * LL, DD, DD);

  // 7) in-place LayerNorm
  ln_kernel<<<NN * LL, 256, 0, stream>>>(out, lng, lnb);
}
